// BiAttnConv_83803401880337
// MI455X (gfx1250) — matrix operand order repaired and checked by reading, not compiled, not run on hardware
//
#include <hip/hip_runtime.h>
#include <math.h>

// ---------------- types ----------------
typedef __bf16  bf16x16 __attribute__((ext_vector_type(16)));
typedef float   f32x8   __attribute__((ext_vector_type(8)));
typedef float   f32x4   __attribute__((ext_vector_type(4)));
typedef unsigned int u32x4 __attribute__((ext_vector_type(4)));

static __device__ __forceinline__ unsigned short f2bf(float f) {
    union { float f; unsigned u; } v; v.f = f;
    unsigned r = v.u + 0x7FFFu + ((v.u >> 16) & 1u);   // round-to-nearest-even
    return (unsigned short)(r >> 16);
}
static __device__ __forceinline__ float bf2f(unsigned short h) {
    union { unsigned u; float f; } v; v.u = ((unsigned)h) << 16;
    return v.f;
}

union FragBF { u32x4 q[2]; bf16x16 v; };

// ---------------- constants ----------------
#define NTOK   12800      // tokens per map (B*H*W = 2*80*80)
#define FDIM   256
#define PDIM   1536
#define HIDDIM 1024
#define HH     80
#define WW     80
#define QK_SCALE 0.17677669529663687f   // (256/8)^-0.5

// =====================================================================
// WMMA bf16 GEMM:  C[M,N] = A[M,K] (bf16) * W[N,:]^T (bf16, row stride ldw)
// Block tile 128x64, 256 threads = 8 waves (4 M-waves x 2 N-waves);
// each wave owns a 32x32 patch = 4 x (16x16x32) WMMA per K-chunk with
// 2 A-fragments and 2 B-fragments (operand reuse).
// Double-buffered LDS staging, software pipelined: next chunk's global
// loads are issued before this chunk's WMMAs; ds_stores after.
// M multiple of 128, N multiple of 64, K multiple of 32 (grids exact),
// no divergence -> EXEC all-ones as required for v_wmma.
// =====================================================================
template <bool RELU, bool OUTBF16>
__global__ __launch_bounds__(256) void gemm_wmma_bf16(
    const unsigned short* __restrict__ A,   // [M,K] bf16
    const unsigned short* __restrict__ W,   // [N, ldw] bf16 (col slice folded into ptr)
    int ldw,
    const float* __restrict__ bias,         // [N]
    void* __restrict__ C, int ldc, int K)
{
    __shared__ alignas(16) unsigned short As[2][128 * 32];
    __shared__ alignas(16) unsigned short Bs[2][64 * 32];

    const int tid  = threadIdx.x;
    const int lane = tid & 31;
    const int wave = tid >> 5;
    const int wm   = wave >> 1;        // 0..3  (M sub-tile, 32 rows each)
    const int wn   = wave & 1;         // 0..1  (N sub-tile, 32 cols each)
    const size_t m0 = (size_t)blockIdx.y * 128;
    const size_t n0 = (size_t)blockIdx.x * 64;

    const int rA = tid >> 2;           // staging row 0..63 (and +64 for A)
    const int cA = (tid & 3) * 8;      // staging k-offset (8 bf16 = 16B)

    const int row = lane & 15;
    const int sel = lane >> 4;         // 0: lanes 0-15, 1: lanes 16-31

    f32x8 c00 = {0.f,0.f,0.f,0.f,0.f,0.f,0.f,0.f};
    f32x8 c01 = c00, c10 = c00, c11 = c00;

    u32x4 ra0, ra1, rb;                // staging registers (pipeline)
    auto gload = [&](int k0) {
        ra0 = *(const u32x4*)(A + (m0 + rA)      * (size_t)K + k0 + cA);
        ra1 = *(const u32x4*)(A + (m0 + rA + 64) * (size_t)K + k0 + cA);
        rb  = *(const u32x4*)(W + (n0 + rA) * (size_t)ldw + k0 + cA);
    };
    auto sstore = [&](int buf) {
        *(u32x4*)(As[buf] + rA * 32 + cA)        = ra0;
        *(u32x4*)(As[buf] + (rA + 64) * 32 + cA) = ra1;
        *(u32x4*)(Bs[buf] + rA * 32 + cA)        = rb;
    };

    gload(0);
    sstore(0);
    int buf = 0;
    for (int k0 = 0; k0 < K; k0 += 32, buf ^= 1) {
        __syncthreads();
        const bool more = (k0 + 32 < K);
        if (more) {
            gload(k0 + 32);            // issue next chunk's VMEM before WMMAs
            __builtin_prefetch(A + (m0 + rA) * (size_t)K + k0 + 64 + cA, 0, 1);
        }

        // A fragments (16x32 bf16): lanes0-15 k-chunks {0,2}, lanes16-31 {1,3}
        FragBF a0, a1, b0, b1;
        const u32x4* Ar0 = (const u32x4*)(As[buf] + (wm * 32 + row) * 32);
        a0.q[0] = Ar0[sel];     a0.q[1] = Ar0[2 + sel];
        const u32x4* Ar1 = (const u32x4*)(As[buf] + (wm * 32 + 16 + row) * 32);
        a1.q[0] = Ar1[sel];     a1.q[1] = Ar1[2 + sel];
        // B fragments (32x16 bf16): lanes0-15 k-chunks {0,2}, lanes16-31 {1,3}
        const u32x4* Br0 = (const u32x4*)(Bs[buf] + (wn * 32 + row) * 32);
        b0.q[0] = Br0[sel]; b0.q[1] = Br0[2 + sel];
        const u32x4* Br1 = (const u32x4*)(Bs[buf] + (wn * 32 + 16 + row) * 32);
        b1.q[0] = Br1[sel]; b1.q[1] = Br1[2 + sel];

        c00 = __builtin_amdgcn_wmma_f32_16x16x32_bf16(false, a0.v, false, b0.v, (short)0, c00, false, false);
        c01 = __builtin_amdgcn_wmma_f32_16x16x32_bf16(false, a0.v, false, b1.v, (short)0, c01, false, false);
        c10 = __builtin_amdgcn_wmma_f32_16x16x32_bf16(false, a1.v, false, b0.v, (short)0, c10, false, false);
        c11 = __builtin_amdgcn_wmma_f32_16x16x32_bf16(false, a1.v, false, b1.v, (short)0, c11, false, false);

        if (more) sstore(buf ^ 1);     // land next chunk into the other buffer
    }

    // epilogue: C/D layout -> lane n = lane%16, M = vgpr + 8*(lane>=16)
    const size_t mb = m0 + wm * 32 + sel * 8;
    const size_t na = n0 + wn * 32 + (lane & 15);
    const float  bz0 = bias[na];
    const float  bz1 = bias[na + 16];
#pragma unroll
    for (int i = 0; i < 8; ++i) {
        const size_t ml = mb + i;        // tiles (m-low, n0/n1)
        const size_t mh = mb + 16 + i;   // tiles (m-high, n0/n1)
        float v00 = c00[i] + bz0;
        float v01 = c01[i] + bz1;
        float v10 = c10[i] + bz0;
        float v11 = c11[i] + bz1;
        if (RELU) {
            v00 = fmaxf(v00, 0.f); v01 = fmaxf(v01, 0.f);
            v10 = fmaxf(v10, 0.f); v11 = fmaxf(v11, 0.f);
        }
        if (OUTBF16) {
            unsigned short* Co = (unsigned short*)C;
            Co[ml * (size_t)ldc + na]      = f2bf(v00);
            Co[ml * (size_t)ldc + na + 16] = f2bf(v01);
            Co[mh * (size_t)ldc + na]      = f2bf(v10);
            Co[mh * (size_t)ldc + na + 16] = f2bf(v11);
        } else {
            float* Co = (float*)C;
            Co[ml * (size_t)ldc + na]      = v00;
            Co[ml * (size_t)ldc + na + 16] = v01;
            Co[mh * (size_t)ldc + na]      = v10;
            Co[mh * (size_t)ldc + na + 16] = v11;
        }
    }
}

// =====================================================================
// 9-tap windowed multi-head attention (head dim 32, zero-padded window):
// one thread per (token, head). Matches the reference's zero-pad gather:
// out-of-bounds taps contribute logit 0 (denominator) and zero value.
// =====================================================================
__global__ __launch_bounds__(256) void attn9_kernel(
    const unsigned short* __restrict__ qb,   // + row*1536 + h*32
    const unsigned short* __restrict__ kb,
    const unsigned short* __restrict__ vb,
    unsigned short* __restrict__ out)        // [NTOK, 256] bf16
{
    const int g = blockIdx.x * 256 + threadIdx.x;   // exactly 12800*8 threads
    const int t = g >> 3;
    const int h = g & 7;
    const int b   = t / (HH * WW);
    const int rem = t - b * (HH * WW);
    const int y   = rem / WW;
    const int x   = rem - y * WW;
    const int d0  = h * 32;

    float q[32];
    const unsigned short* qp = qb + (size_t)t * PDIM + d0;
#pragma unroll
    for (int d = 0; d < 32; ++d) q[d] = bf2f(qp[d]) * QK_SCALE;

    float lg[9];
#pragma unroll
    for (int tap = 0; tap < 9; ++tap) {
        const int yy = y + tap / 3 - 1;
        const int xx = x + tap % 3 - 1;
        float l = 0.f;
        if (yy >= 0 && yy < HH && xx >= 0 && xx < WW) {
            const unsigned short* kp =
                kb + ((size_t)(b * HH * WW + yy * WW + xx)) * PDIM + d0;
#pragma unroll
            for (int d = 0; d < 32; ++d) l += q[d] * bf2f(kp[d]);
        }
        lg[tap] = l;
    }
    float mx = lg[0];
#pragma unroll
    for (int tap = 1; tap < 9; ++tap) mx = fmaxf(mx, lg[tap]);
    float den = 0.f;
#pragma unroll
    for (int tap = 0; tap < 9; ++tap) { lg[tap] = __expf(lg[tap] - mx); den += lg[tap]; }
    const float inv = 1.f / den;

    float acc[32];
#pragma unroll
    for (int d = 0; d < 32; ++d) acc[d] = 0.f;
#pragma unroll
    for (int tap = 0; tap < 9; ++tap) {
        const int yy = y + tap / 3 - 1;
        const int xx = x + tap % 3 - 1;
        if (yy >= 0 && yy < HH && xx >= 0 && xx < WW) {
            const unsigned short* vp =
                vb + ((size_t)(b * HH * WW + yy * WW + xx)) * PDIM + d0;
            const float wv = lg[tap] * inv;
#pragma unroll
            for (int d = 0; d < 32; ++d) acc[d] += wv * bf2f(vp[d]);
        }
    }
    unsigned short* op = out + (size_t)t * FDIM + d0;
#pragma unroll
    for (int d = 0; d < 32; ++d) op[d] = f2bf(acc[d]);
}

// =====================================================================
// Residual add (2 or 3 operands) + LayerNorm over 256 features.
// One wave32 per token; 8 features per lane; shfl_xor tree reductions.
// Optionally also emits bf16 copy (next GEMM's A operand).
// =====================================================================
__global__ __launch_bounds__(256) void add_ln_kernel(
    const float* __restrict__ xin, const float* __restrict__ a,
    const float* __restrict__ bopt,           // may be nullptr
    const float* __restrict__ g, const float* __restrict__ bt,
    float* __restrict__ out, unsigned short* __restrict__ out_bf)
{
    const int t    = blockIdx.x * 8 + (threadIdx.x >> 5);
    const int lane = threadIdx.x & 31;
    const size_t base = (size_t)t * FDIM;

    float v[8];
    float s = 0.f;
#pragma unroll
    for (int j = 0; j < 8; ++j) {
        const int f = j * 32 + lane;
        float x = xin[base + f] + a[base + f];
        if (bopt) x += bopt[base + f];
        v[j] = x; s += x;
    }
#pragma unroll
    for (int o = 16; o > 0; o >>= 1) s += __shfl_xor(s, o, 32);
    const float mean = s * (1.f / 256.f);

    float qs = 0.f;
#pragma unroll
    for (int j = 0; j < 8; ++j) { v[j] -= mean; qs += v[j] * v[j]; }
#pragma unroll
    for (int o = 16; o > 0; o >>= 1) qs += __shfl_xor(qs, o, 32);
    const float inv = rsqrtf(qs * (1.f / 256.f) + 1e-5f);

#pragma unroll
    for (int j = 0; j < 8; ++j) {
        const int f = j * 32 + lane;
        const float yv = g[f] * v[j] * inv + bt[f];
        out[base + f] = yv;
        if (out_bf) out_bf[base + f] = f2bf(yv);
    }
}

// ---------------- f32 -> bf16 conversion (4 elems/thread) ----------------
__global__ __launch_bounds__(256) void cvt_bf16_kernel(
    const float* __restrict__ in, unsigned short* __restrict__ out, int n4)
{
    const int i = blockIdx.x * 256 + threadIdx.x;
    if (i >= n4) return;
    f32x4 f = ((const f32x4*)in)[i];
    out[i * 4 + 0] = f2bf(f[0]);
    out[i * 4 + 1] = f2bf(f[1]);
    out[i * 4 + 2] = f2bf(f[2]);
    out[i * 4 + 3] = f2bf(f[3]);
}

// =====================================================================
// host launch
// =====================================================================
extern "C" void kernel_launch(void* const* d_in, const int* in_sizes, int n_in,
                              void* d_out, int out_size, void* d_ws, size_t ws_size,
                              hipStream_t stream) {
    (void)in_sizes; (void)n_in; (void)out_size; (void)ws_size;

    const float* feat0   = (const float*)d_in[0];
    const float* feat1   = (const float*)d_in[1];
    const float* in_w0   = (const float*)d_in[2];
    const float* in_b0   = (const float*)d_in[3];
    const float* in_w1   = (const float*)d_in[4];
    const float* in_b1   = (const float*)d_in[5];
    const float* out_w0  = (const float*)d_in[6];
    const float* out_b0  = (const float*)d_in[7];
    const float* out_w1  = (const float*)d_in[8];
    const float* out_b1  = (const float*)d_in[9];
    const float* ln1_g0  = (const float*)d_in[10];
    const float* ln1_b0  = (const float*)d_in[11];
    const float* ln1_g1  = (const float*)d_in[12];
    const float* ln1_b1  = (const float*)d_in[13];
    const float* up_w0   = (const float*)d_in[14];
    const float* up_b0   = (const float*)d_in[15];
    const float* dn_w0   = (const float*)d_in[16];
    const float* dn_b0   = (const float*)d_in[17];
    const float* up_w1   = (const float*)d_in[18];
    const float* up_b1   = (const float*)d_in[19];
    const float* dn_w1   = (const float*)d_in[20];
    const float* dn_b1   = (const float*)d_in[21];
    const float* ln2_g0  = (const float*)d_in[22];
    const float* ln2_b0  = (const float*)d_in[23];
    const float* ln2_g1  = (const float*)d_in[24];
    const float* ln2_b1  = (const float*)d_in[25];
    float* out = (float*)d_out;

    // ---- bump allocator over workspace (~253 MB needed) ----
    char*  ws  = (char*)d_ws;
    size_t off = 0;
    auto alloc = [&](size_t bytes) -> void* {
        void* p = ws + off;
        off += (bytes + 255) & ~(size_t)255;
        return p;
    };
    typedef unsigned short u16;
    const size_t NT = NTOK;

    u16* feat0_bf = (u16*)alloc(NT * FDIM * 2);
    u16* feat1_bf = (u16*)alloc(NT * FDIM * 2);
    u16* w_in0    = (u16*)alloc((size_t)PDIM * FDIM * 2);
    u16* w_in1    = (u16*)alloc((size_t)PDIM * FDIM * 2);
    u16* w_out0   = (u16*)alloc((size_t)FDIM * 3 * FDIM * 2);
    u16* w_out1   = (u16*)alloc((size_t)FDIM * 3 * FDIM * 2);
    u16* w_up0    = (u16*)alloc((size_t)HIDDIM * FDIM * 2);
    u16* w_up1    = (u16*)alloc((size_t)HIDDIM * FDIM * 2);
    u16* w_dn0    = (u16*)alloc((size_t)FDIM * HIDDIM * 2);
    u16* w_dn1    = (u16*)alloc((size_t)FDIM * HIDDIM * 2);
    u16* p0       = (u16*)alloc(NT * PDIM * 2);
    u16* p1       = (u16*)alloc(NT * PDIM * 2);
    u16* att_sa0  = (u16*)alloc(NT * FDIM * 2);
    u16* att_sa1  = (u16*)alloc(NT * FDIM * 2);
    u16* att_td   = (u16*)alloc(NT * FDIM * 2);
    u16* att_bu   = (u16*)alloc(NT * FDIM * 2);
    float* sa0p   = (float*)alloc(NT * FDIM * 4);
    float* sa1p   = (float*)alloc(NT * FDIM * 4);
    float* td0p   = (float*)alloc(NT * FDIM * 4);
    float* bu0p   = (float*)alloc(NT * FDIM * 4);
    float* x0     = (float*)alloc(NT * FDIM * 4);
    float* x1     = (float*)alloc(NT * FDIM * 4);
    u16* x0_bf    = (u16*)alloc(NT * FDIM * 2);
    u16* x1_bf    = (u16*)alloc(NT * FDIM * 2);
    u16* h_bf     = (u16*)alloc(NT * HIDDIM * 2);   // reused for both maps
    float* ybuf   = (float*)alloc(NT * FDIM * 4);   // reused for both maps

    auto cvt = [&](const float* src, u16* dst, size_t n) {
        const int n4 = (int)(n / 4);
        cvt_bf16_kernel<<<(n4 + 255) / 256, 256, 0, stream>>>(src, dst, n4);
    };

    // ---- 1) bf16 conversions (activations + weights) ----
    cvt(feat0, feat0_bf, NT * FDIM);
    cvt(feat1, feat1_bf, NT * FDIM);
    cvt(in_w0, w_in0, (size_t)PDIM * FDIM);
    cvt(in_w1, w_in1, (size_t)PDIM * FDIM);
    cvt(out_w0, w_out0, (size_t)FDIM * 3 * FDIM);
    cvt(out_w1, w_out1, (size_t)FDIM * 3 * FDIM);
    cvt(up_w0, w_up0, (size_t)HIDDIM * FDIM);
    cvt(up_w1, w_up1, (size_t)HIDDIM * FDIM);
    cvt(dn_w0, w_dn0, (size_t)FDIM * HIDDIM);
    cvt(dn_w1, w_dn1, (size_t)FDIM * HIDDIM);

    const dim3 blk(256);
    const int MB = (int)(NT / 128);   // 100 block-rows

    // ---- 2) in-projection: p = feat @ in_w^T + in_b  (N=1536, K=256) ----
    gemm_wmma_bf16<false, true><<<dim3(PDIM / 64, MB), blk, 0, stream>>>(
        feat0_bf, w_in0, FDIM, in_b0, p0, PDIM, FDIM);
    gemm_wmma_bf16<false, true><<<dim3(PDIM / 64, MB), blk, 0, stream>>>(
        feat1_bf, w_in1, FDIM, in_b1, p1, PDIM, FDIM);

    // ---- 3) four 9-tap attentions ----
    const int ATT_BLKS = (int)(NT * 8 / 256);   // 400
    attn9_kernel<<<ATT_BLKS, blk, 0, stream>>>(p0,             p0 + FDIM,      p0 + 2 * FDIM, att_sa0);
    attn9_kernel<<<ATT_BLKS, blk, 0, stream>>>(p1,             p1 + FDIM,      p1 + 2 * FDIM, att_sa1);
    attn9_kernel<<<ATT_BLKS, blk, 0, stream>>>(p0 + 3 * FDIM,  p1 + 4 * FDIM,  p1 + 5 * FDIM, att_td);
    attn9_kernel<<<ATT_BLKS, blk, 0, stream>>>(p1 + 3 * FDIM,  p0 + 4 * FDIM,  p0 + 5 * FDIM, att_bu);

    // ---- 4) output projections (N=256, K=256) ----
    gemm_wmma_bf16<false, false><<<dim3(FDIM / 64, MB), blk, 0, stream>>>(
        att_sa0, w_out0,        3 * FDIM, out_b0,        sa0p, FDIM, FDIM);
    gemm_wmma_bf16<false, false><<<dim3(FDIM / 64, MB), blk, 0, stream>>>(
        att_sa1, w_out1,        3 * FDIM, out_b1,        sa1p, FDIM, FDIM);
    gemm_wmma_bf16<false, false><<<dim3(FDIM / 64, MB), blk, 0, stream>>>(
        att_td,  w_out0 + FDIM, 3 * FDIM, out_b0 + FDIM, td0p, FDIM, FDIM);
    gemm_wmma_bf16<false, false><<<dim3(FDIM / 64, MB), blk, 0, stream>>>(
        att_bu,  w_out0 + FDIM, 3 * FDIM, out_b0 + FDIM, bu0p, FDIM, FDIM);

    // ---- 5) residual + LN1 ----
    const int LN_BLKS = (int)(NT / 8);   // 1600
    add_ln_kernel<<<LN_BLKS, blk, 0, stream>>>(feat0, sa0p, td0p, ln1_g0, ln1_b0, x0, x0_bf);
    add_ln_kernel<<<LN_BLKS, blk, 0, stream>>>(feat1, sa1p, bu0p, ln1_g1, ln1_b1, x1, x1_bf);

    // ---- 6) FFN + LN2, map 0 ----
    gemm_wmma_bf16<true, true><<<dim3(HIDDIM / 64, MB), blk, 0, stream>>>(
        x0_bf, w_up0, FDIM, up_b0, h_bf, HIDDIM, FDIM);
    gemm_wmma_bf16<false, false><<<dim3(FDIM / 64, MB), blk, 0, stream>>>(
        h_bf, w_dn0, HIDDIM, dn_b0, ybuf, FDIM, HIDDIM);
    add_ln_kernel<<<LN_BLKS, blk, 0, stream>>>(x0, ybuf, nullptr, ln2_g0, ln2_b0,
                                               out, nullptr);

    // ---- 7) FFN + LN2, map 1 ----
    gemm_wmma_bf16<true, true><<<dim3(HIDDIM / 64, MB), blk, 0, stream>>>(
        x1_bf, w_up1, FDIM, up_b1, h_bf, HIDDIM, FDIM);
    gemm_wmma_bf16<false, false><<<dim3(FDIM / 64, MB), blk, 0, stream>>>(
        h_bf, w_dn1, HIDDIM, dn_b1, ybuf, FDIM, HIDDIM);
    add_ln_kernel<<<LN_BLKS, blk, 0, stream>>>(x1, ybuf, nullptr, ln2_g1, ln2_b1,
                                               out + NT * FDIM, nullptr);
}
